// MixerDiffAttention_81063212745407
// MI455X (gfx1250) — matrix operand order repaired and checked by reading, not compiled, not run on hardware
//
#include <hip/hip_runtime.h>
#include <hip/hip_bf16.h>

typedef __attribute__((ext_vector_type(16))) __bf16       v16bf;
typedef __attribute__((ext_vector_type(8)))  float        v8f;
typedef __attribute__((ext_vector_type(4)))  unsigned int u32x4;
typedef __attribute__((address_space(3)))    char         lds_char;

union Frag { v16bf v; u32x4 q[2]; };

static __device__ inline v8f wmma_bf16(v16bf a, v16bf b, v8f c) {
  return __builtin_amdgcn_wmma_f32_16x16x32_bf16(
      /*neg_a=*/false, a, /*neg_b=*/false, b,
      /*c_mod=*/(short)0, c, /*reuse_a=*/false, /*reuse_b=*/false);
}

// Async global->LDS copy of 16 bytes per lane (ASYNCcnt-tracked, bypasses VGPRs).
static __device__ inline void async_copy_b128(const __bf16* g, __bf16* l) {
  unsigned la = (unsigned)(size_t)(lds_char*)l;            // LDS byte offset
  unsigned long long ga = (unsigned long long)g;           // 64-bit global addr
  asm volatile("global_load_async_to_lds_b128 %0, %1, off"
               :: "v"(la), "v"(ga) : "memory");
}
static __device__ inline void wait_async_le4() {
  asm volatile("s_wait_asynccnt 0x4" ::: "memory");
}
static __device__ inline void wait_async_0() {
  asm volatile("s_wait_asynccnt 0x0" ::: "memory");
}

// ---------------------------------------------------------------- convert
__global__ __launch_bounds__(256) void cvt_kernel(const float* __restrict__ in,
                                                  __bf16* __restrict__ out, int n) {
  int i = blockIdx.x * 256 + threadIdx.x;
  if (i < n) out[i] = (__bf16)in[i];
}

// ---------------------------------------------------------------- GEMM
// C[M][N] = A[M][K] * Bt[N][K]^T   (A row-major bf16, Bt row-major bf16, C f32)
// WG tile 128x128, 8 waves each 64(M)x32(N), K step 32, double-buffered LDS
// filled by async global->LDS DMA; last K-step peeled so the hot loop is
// branch-free: issue-DMA / s_wait_asynccnt 4 / barrier / 8 WMMAs / barrier.
__global__ __launch_bounds__(256) void gemm_bf16(const __bf16* __restrict__ A,
                                                 const __bf16* __restrict__ Bt,
                                                 float* __restrict__ C,
                                                 int M, int N, int K) {
  __shared__ __bf16 aS[2][128][32];
  __shared__ __bf16 bS[2][128][32];
  const int tid = threadIdx.x;
  const int w = tid >> 5, lane = tid & 31, r = lane & 15, half = lane >> 4;
  const int m0 = blockIdx.x * 128, n0 = blockIdx.y * 128;
  const int wm = (w >> 2) * 64, wn = (w & 3) * 32;
  v8f z = {0.f,0.f,0.f,0.f,0.f,0.f,0.f,0.f};
  v8f acc[4][2];
  for (int i = 0; i < 4; ++i) for (int j = 0; j < 2; ++j) acc[i][j] = z;

  const int lr = tid >> 1, lc = (tid & 1) * 16;
  auto prefetch = [&](int buf, int k0) {
    const __bf16* ag = A  + (size_t)(m0 + lr) * K + k0 + lc;
    const __bf16* bg = Bt + (size_t)(n0 + lr) * K + k0 + lc;
    async_copy_b128(ag,     &aS[buf][lr][lc]);
    async_copy_b128(ag + 8, &aS[buf][lr][lc + 8]);
    async_copy_b128(bg,     &bS[buf][lr][lc]);
    async_copy_b128(bg + 8, &bS[buf][lr][lc + 8]);
  };
  auto compute = [&](int cur) {
    Frag af[4], bfr[2];
#pragma unroll
    for (int mt = 0; mt < 4; ++mt) {
      const __bf16* ap = &aS[cur][wm + mt * 16 + r][0];
      af[mt].q[0] = *(const u32x4*)(ap + half * 8);
      af[mt].q[1] = *(const u32x4*)(ap + 16 + half * 8);
    }
#pragma unroll
    for (int nt = 0; nt < 2; ++nt) {
      const __bf16* bp = &bS[cur][wn + nt * 16 + r][0];
      bfr[nt].q[0] = *(const u32x4*)(bp + half * 8);
      bfr[nt].q[1] = *(const u32x4*)(bp + 16 + half * 8);
    }
#pragma unroll
    for (int mt = 0; mt < 4; ++mt)
#pragma unroll
      for (int nt = 0; nt < 2; ++nt)
        acc[mt][nt] = wmma_bf16(af[mt].v, bfr[nt].v, acc[mt][nt]);
  };

  const int nsteps = K >> 5;
  prefetch(0, 0);
  // steady state: branch-free hot loop
  for (int i = 0; i < nsteps - 1; ++i) {
    const int cur = i & 1;
    prefetch(cur ^ 1, (i + 1) * 32);   // overlap next K-step DMA with compute
    wait_async_le4();                  // cur's 4 DMAs done (in-order completion)
    __syncthreads();
    compute(cur);
    __syncthreads();                   // all reads done before buffer reuse
  }
  // peeled final K-step
  wait_async_0();
  __syncthreads();
  compute((nsteps - 1) & 1);

#pragma unroll
  for (int mt = 0; mt < 4; ++mt)
#pragma unroll
    for (int nt = 0; nt < 2; ++nt)
#pragma unroll
      for (int v = 0; v < 8; ++v) {
        int row = m0 + wm + mt * 16 + v + 8 * half;
        int col = n0 + wn + nt * 16 + r;
        C[(size_t)row * N + col] = acc[mt][nt][v];
      }
}

// ------------------------------------------------- RMSNorm + rotary + pack
// In: QK f32 [T][1024] viewed as (T,2,8,64). Out: [h][t][64] bf16 (x2 attn).
__global__ __launch_bounds__(256) void qkpack_kernel(const float* __restrict__ X,
                                                     __bf16* __restrict__ o1,
                                                     __bf16* __restrict__ o2,
                                                     float scale, int T) {
  const int w = threadIdx.x >> 5, lane = threadIdx.x & 31;
  const int gw = blockIdx.x * 8 + w;           // T*2*8 waves total
  const int t = gw >> 4, a = (gw >> 3) & 1, h = gw & 7;
  const float* src = X + (size_t)t * 1024 + a * 512 + h * 64;
  float v1 = src[lane], v2 = src[lane + 32];
  float ss = v1 * v1 + v2 * v2;
#pragma unroll
  for (int mk = 1; mk < 32; mk <<= 1) ss += __shfl_xor(ss, mk, 32);
  float rn = rsqrtf(ss * (1.0f / 64.0f) + 1.1920929e-7f);
  float x1 = v1 * rn, x2 = v2 * rn;
  float fr = (float)t * __powf(10000.0f, -(float)lane * (1.0f / 32.0f));
  float s, c;
  __sincosf(fr, &s, &c);
  __bf16* dst = (a ? o2 : o1) + ((size_t)h * T + t) * 64;
  dst[lane]      = (__bf16)((x1 * c + x2 * s) * scale);
  dst[lane + 32] = (__bf16)((-x1 * s + x2 * c) * scale);
}

// V f32 [t][h*128+d] -> Vt bf16 [h][d][t]  (B-fragment friendly layout)
__global__ __launch_bounds__(256) void vpack_kernel(const float* __restrict__ V,
                                                    __bf16* __restrict__ Vt, int T) {
  int idx = blockIdx.x * 256 + threadIdx.x;
  int t = idx >> 10, c = idx & 1023;
  int h = c >> 7, d = c & 127;
  Vt[((size_t)h * 128 + d) * T + t] = (__bf16)V[idx];
}

// ---------------------------------------------------------------- lambda
__global__ void lambda_kernel(const float* lq1, const float* lk1,
                              const float* lq2, const float* lk2, float* lam) {
  int i = threadIdx.x;  // 32 lanes, HEAD_DIM/2 == 32
  float d1 = lq1[i] * lk1[i], d2 = lq2[i] * lk2[i];
#pragma unroll
  for (int mk = 1; mk < 32; mk <<= 1) {
    d1 += __shfl_xor(d1, mk, 32);
    d2 += __shfl_xor(d2, mk, 32);
  }
  if (i == 0) *lam = __expf(d1) - __expf(d2) + 0.35550906f;  // + LAMBDA_INIT
}

// ------------------------------------------------------------ flash attn
// grid (T/128, 8 heads, 2 attns); 8 waves, each wave owns 16 query rows.
__global__ __launch_bounds__(256) void attn_kernel(const __bf16* __restrict__ Q1,
                                                   const __bf16* __restrict__ Q2,
                                                   const __bf16* __restrict__ K1,
                                                   const __bf16* __restrict__ K2,
                                                   const __bf16* __restrict__ Vt,
                                                   float* __restrict__ Y, int T) {
  __shared__ __bf16 pstage[8][16][32];
  const int w = threadIdx.x >> 5, lane = threadIdx.x & 31;
  const int r = lane & 15, half = lane >> 4;
  const int t0 = (blockIdx.x * 8 + w) * 16;
  const int h = blockIdx.y;
  const __bf16* Qs = blockIdx.z ? Q2 : Q1;
  const __bf16* Ks = blockIdx.z ? K2 : K1;
  const __bf16* Qh = Qs + (size_t)h * T * 64;
  const __bf16* Kh = Ks + (size_t)h * T * 64;
  const __bf16* Vh = Vt + (size_t)h * 128 * T;
  float* Yp = Y + (size_t)blockIdx.z * T * 1024;

  Frag qf[2];
  {
    const __bf16* qrow = Qh + (size_t)(t0 + r) * 64;
    qf[0].q[0] = *(const u32x4*)(qrow + half * 8);
    qf[0].q[1] = *(const u32x4*)(qrow + 16 + half * 8);
    qf[1].q[0] = *(const u32x4*)(qrow + 32 + half * 8);
    qf[1].q[1] = *(const u32x4*)(qrow + 48 + half * 8);
  }
  v8f z = {0.f,0.f,0.f,0.f,0.f,0.f,0.f,0.f};
  v8f o[8];
  float m[8], l[8];
#pragma unroll
  for (int i = 0; i < 8; ++i) { o[i] = z; m[i] = -3.0e38f; l[i] = 0.f; }

  __bf16* pb = &pstage[w][0][0];
  for (int s0 = 0; s0 < t0 + 16; s0 += 32) {
    // prefetch next s-block's K rows (1/lane) and V stripes (4/lane) into L2/L0
    if (s0 + 32 < t0 + 16) {
      __builtin_prefetch(Kh + (size_t)(s0 + 32 + lane) * 64, 0, 0);
#pragma unroll
      for (int j = 0; j < 4; ++j)
        __builtin_prefetch(Vh + (size_t)(j * 32 + lane) * T + s0 + 32, 0, 0);
    }
    // ---- scores: 16 x 32 block, K-dim 64 = 2 WMMA per 16x16 tile
    v8f sc[2]; sc[0] = z; sc[1] = z;
#pragma unroll
    for (int st = 0; st < 2; ++st)
#pragma unroll
      for (int kd = 0; kd < 2; ++kd) {
        Frag bf;
        const __bf16* krow = Kh + (size_t)(s0 + st * 16 + r) * 64 + kd * 32 + half * 8;
        bf.q[0] = *(const u32x4*)(krow);
        bf.q[1] = *(const u32x4*)(krow + 16);
        sc[st] = wmma_bf16(qf[kd].v, bf.v, sc[st]);
      }
    // ---- causal mask (diagonal blocks)
    if (s0 + 32 > t0) {
#pragma unroll
      for (int st = 0; st < 2; ++st)
#pragma unroll
        for (int v = 0; v < 8; ++v) {
          int s = s0 + st * 16 + r, t = t0 + v + 8 * half;
          if (s > t) sc[st][v] = -3.0e38f;
        }
    }
    // ---- online softmax: row max / sum via 16-lane half-wave shuffles
#pragma unroll
    for (int v = 0; v < 8; ++v) {
      float rm = fmaxf(sc[0][v], sc[1][v]);
#pragma unroll
      for (int mk = 1; mk < 16; mk <<= 1) rm = fmaxf(rm, __shfl_xor(rm, mk, 32));
      float mn = fmaxf(m[v], rm);
      float p0 = __expf(sc[0][v] - mn);
      float p1 = __expf(sc[1][v] - mn);
      pb[(v + 8 * half) * 32 + r]      = (__bf16)p0;
      pb[(v + 8 * half) * 32 + 16 + r] = (__bf16)p1;
      float rs = p0 + p1;
#pragma unroll
      for (int mk = 1; mk < 16; mk <<= 1) rs += __shfl_xor(rs, mk, 32);
      float alpha = __expf(m[v] - mn);
      m[v] = mn;
      l[v] = l[v] * alpha + rs;
#pragma unroll
      for (int dt = 0; dt < 8; ++dt) o[dt][v] *= alpha;
    }
    // LDS in-order per wave; force write->read ordering before A-frag reload
    asm volatile("" ::: "memory");
    asm volatile("s_wait_dscnt 0x0" ::: "memory");
    Frag pf;
    pf.q[0] = *(const u32x4*)(pb + r * 32 + half * 8);
    pf.q[1] = *(const u32x4*)(pb + r * 32 + 16 + half * 8);
    // ---- O += P * V   (8 d-tiles of 16, K = 32 keys)
#pragma unroll
    for (int dt = 0; dt < 8; ++dt) {
      Frag vf;
      const __bf16* vrow = Vh + (size_t)(dt * 16 + r) * T + s0 + half * 8;
      vf.q[0] = *(const u32x4*)(vrow);
      vf.q[1] = *(const u32x4*)(vrow + 16);
      o[dt] = wmma_bf16(pf.v, vf.v, o[dt]);
    }
  }
#pragma unroll
  for (int v = 0; v < 8; ++v) {
    float inv = 1.0f / l[v];
    int t = t0 + v + 8 * half;
#pragma unroll
    for (int dt = 0; dt < 8; ++dt)
      Yp[(size_t)t * 1024 + h * 128 + dt * 16 + r] = o[dt][v] * inv;
  }
}

// ---------------------------------------------------------------- combine
__global__ __launch_bounds__(256) void combine_kernel(const float* __restrict__ Y,
                                                      const float* __restrict__ lam,
                                                      __bf16* __restrict__ yc, int n) {
  int i = blockIdx.x * 256 + threadIdx.x;
  float lf = *lam;
  yc[i] = (__bf16)(Y[i] - lf * Y[i + n]);
}

// ================================================================ launch
extern "C" void kernel_launch(void* const* d_in, const int* in_sizes, int n_in,
                              void* d_out, int out_size, void* d_ws, size_t ws_size,
                              hipStream_t stream) {
  (void)in_sizes; (void)n_in; (void)out_size; (void)ws_size;
  const float* x   = (const float*)d_in[0];
  const float* Wq  = (const float*)d_in[1];
  const float* Wk  = (const float*)d_in[2];
  const float* Wv  = (const float*)d_in[3];
  const float* Wp  = (const float*)d_in[4];
  const float* lq1 = (const float*)d_in[5];
  const float* lk1 = (const float*)d_in[6];
  const float* lq2 = (const float*)d_in[7];
  const float* lk2 = (const float*)d_in[8];
  const int T = 2048, D = 1024;

  char* p = (char*)d_ws;
  auto carve = [&](size_t bytes) {
    char* r = p;
    p += (bytes + 255) & ~(size_t)255;
    return r;
  };
  __bf16* xb  = (__bf16*)carve((size_t)T * D * 2);
  __bf16* wqb = (__bf16*)carve((size_t)D * D * 2);
  __bf16* wkb = (__bf16*)carve((size_t)D * D * 2);
  __bf16* wvb = (__bf16*)carve((size_t)D * D * 2);
  __bf16* wpb = (__bf16*)carve((size_t)D * D * 2);
  float*  Qf  = (float*)carve((size_t)T * D * 4);
  float*  Kf  = (float*)carve((size_t)T * D * 4);
  float*  Vf  = (float*)carve((size_t)T * D * 4);
  __bf16* q1  = (__bf16*)carve((size_t)8 * T * 64 * 2);
  __bf16* q2  = (__bf16*)carve((size_t)8 * T * 64 * 2);
  __bf16* k1  = (__bf16*)carve((size_t)8 * T * 64 * 2);
  __bf16* k2  = (__bf16*)carve((size_t)8 * T * 64 * 2);
  __bf16* vt  = (__bf16*)carve((size_t)8 * 128 * T * 2);
  float*  Y   = (float*)carve((size_t)2 * T * D * 4);
  __bf16* yc  = (__bf16*)carve((size_t)T * D * 2);
  float*  lam = (float*)carve(256);

  cvt_kernel<<<(T * D) / 256, 256, 0, stream>>>(x, xb, T * D);
  cvt_kernel<<<(D * D) / 256, 256, 0, stream>>>(Wq, wqb, D * D);
  cvt_kernel<<<(D * D) / 256, 256, 0, stream>>>(Wk, wkb, D * D);
  cvt_kernel<<<(D * D) / 256, 256, 0, stream>>>(Wv, wvb, D * D);
  cvt_kernel<<<(D * D) / 256, 256, 0, stream>>>(Wp, wpb, D * D);

  dim3 gg(T / 128, D / 128);
  gemm_bf16<<<gg, 256, 0, stream>>>(xb, wqb, Qf, T, D, D);
  gemm_bf16<<<gg, 256, 0, stream>>>(xb, wkb, Kf, T, D, D);
  gemm_bf16<<<gg, 256, 0, stream>>>(xb, wvb, Vf, T, D, D);

  qkpack_kernel<<<(T * 16) / 8, 256, 0, stream>>>(Qf, q1, q2, 0.125f, T);
  qkpack_kernel<<<(T * 16) / 8, 256, 0, stream>>>(Kf, k1, k2, 1.0f, T);
  vpack_kernel<<<(T * D) / 256, 256, 0, stream>>>(Vf, vt, T);
  lambda_kernel<<<1, 32, 0, stream>>>(lq1, lk1, lq2, lk2, lam);

  attn_kernel<<<dim3(T / 128, 8, 2), 256, 0, stream>>>(q1, q2, k1, k2, vt, Y, T);

  combine_kernel<<<(T * D) / 256, 256, 0, stream>>>(Y, lam, yc, T * D);
  gemm_bf16<<<gg, 256, 0, stream>>>(yc, wpb, (float*)d_out, T, D, D);
}
